// MSA_996432413274
// MI455X (gfx1250) — compile-verified, matrix operand order checked by reading, not run on hardware
//
#include <hip/hip_runtime.h>
#include <hip/hip_bf16.h>

typedef __attribute__((ext_vector_type(16))) __bf16 v16bf;
typedef __attribute__((ext_vector_type(8)))  float  v8f;

#define BS 2
#define SEQ 4096
#define DMODEL 512
#define NHEAD 8
#define DHEAD 64
#define E3 (3 * DMODEL)   // 1536
#define MTOT (BS * SEQ)   // 8192

// ---------------------------------------------------------------------------
// WMMA helper: D = A(16x32 bf16) * B(32x16 bf16) + C(16x16 f32)
// ---------------------------------------------------------------------------
__device__ __forceinline__ v8f wmma_bf16(v16bf a, v16bf b, v8f c) {
    return __builtin_amdgcn_wmma_f32_16x16x32_bf16(
        /*neg_a=*/false, a, /*neg_b=*/false, b,
        /*c_mod=*/(short)0, c, /*reuse_a=*/false, /*reuse_b=*/false);
}

// A/B-fragment with K contiguous in memory per lane:
// lanes 0-15 -> row=lane, K in {0..7,16..23}; lanes 16-31 -> row=lane-16,
// K in {8..15,24..31}. Vectorizes to two 16B loads per lane.
__device__ __forceinline__ v16bf load_frag_kcontig(const __bf16* base, int ld) {
    const int lane = threadIdx.x & 31;
    const int row  = lane & 15;
    const int khi  = (lane >> 4) ? 8 : 0;
    const __bf16* p = base + (size_t)row * ld;
    v16bf f;
#pragma unroll
    for (int v = 0; v < 8; ++v) {
        const int k = (v < 4) ? (khi + 2 * v) : (16 + khi + 2 * (v - 4));
        f[2 * v]     = p[k];
        f[2 * v + 1] = p[k + 1];
    }
    return f;
}

// CDNA5 async copy: global -> LDS, 16 bytes per lane, tracked by ASYNCcnt.
__device__ __forceinline__ void async_load_b128(unsigned lds_off, const void* gbase,
                                                unsigned byte_off) {
    asm volatile("global_load_async_to_lds_b128 %0, %1, %2"
                 :: "v"(lds_off), "v"(byte_off), "s"(gbase) : "memory");
}
__device__ __forceinline__ void wait_asynccnt0() {
    asm volatile("s_wait_asynccnt 0" ::: "memory");
}
__device__ __forceinline__ unsigned lds_offset_of(const void* p) {
    // generic LDS address = {SHARED_BASE, offset[31:0]} -> low 32 bits are offset
    return (unsigned)(unsigned long long)p;
}

// ---------------------------------------------------------------------------
// fp32 -> bf16 streaming convert
// ---------------------------------------------------------------------------
__global__ __launch_bounds__(256) void cvt_f32_bf16(const float* __restrict__ s,
                                                    __bf16* __restrict__ d, long n) {
    long i = (long)blockIdx.x * blockDim.x + threadIdx.x;
    long stride = (long)gridDim.x * blockDim.x;
    for (; i < n; i += stride) d[i] = (__bf16)s[i];
}

// ---------------------------------------------------------------------------
// QKV projection: qkv[m,e] = sum_d z[m,d] * w_qkv[e,d]
// wave tile: 32 rows x 64 cols (8 C-frags); block = 8 waves = 256 rows
// ---------------------------------------------------------------------------
__global__ __launch_bounds__(256) void qkv_gemm(const __bf16* __restrict__ zb,
                                                const __bf16* __restrict__ wb,
                                                __bf16* __restrict__ q,
                                                __bf16* __restrict__ k,
                                                __bf16* __restrict__ v) {
    const int wave = threadIdx.x >> 5;
    const int m0 = blockIdx.x * 256 + wave * 32;
    const int e0 = blockIdx.y * 64;

    v8f acc[2][4];
#pragma unroll
    for (int i = 0; i < 2; ++i)
#pragma unroll
        for (int t = 0; t < 4; ++t) acc[i][t] = v8f{};

    for (int kk = 0; kk < DMODEL; kk += 32) {
        v16bf a0 = load_frag_kcontig(zb + (size_t)m0 * DMODEL + kk, DMODEL);
        v16bf a1 = load_frag_kcontig(zb + (size_t)(m0 + 16) * DMODEL + kk, DMODEL);
#pragma unroll
        for (int t = 0; t < 4; ++t) {
            v16bf b = load_frag_kcontig(wb + (size_t)(e0 + 16 * t) * DMODEL + kk, DMODEL);
            acc[0][t] = wmma_bf16(a0, b, acc[0][t]);
            acc[1][t] = wmma_bf16(a1, b, acc[1][t]);
        }
    }

    const int lane  = threadIdx.x & 31;
    const int ncol  = lane & 15;
    const int rbase = (lane >> 4) ? 8 : 0;
#pragma unroll
    for (int i = 0; i < 2; ++i) {
#pragma unroll
        for (int t = 0; t < 4; ++t) {
#pragma unroll
            for (int r = 0; r < 8; ++r) {
                const int m = m0 + 16 * i + rbase + r;
                const int e = e0 + 16 * t + ncol;
                const int which = e >> 9;          // 0=q 1=k 2=v
                const int rem   = e & (DMODEL - 1);
                const int h     = rem >> 6;
                const int dh    = rem & (DHEAD - 1);
                const int b_    = m >> 12;
                const int nrow  = m & (SEQ - 1);
                __bf16* dst = (which == 0) ? q : (which == 1) ? k : v;
                dst[((((size_t)b_ * NHEAD + h) * SEQ) + nrow) * DHEAD + dh] =
                    (__bf16)acc[i][t][r];
            }
        }
    }
}

// ---------------------------------------------------------------------------
// Flash attention with LDS staging:
//  - K tile (32x64) staged via global_load_async_to_lds_b128 (ASYNCcnt path)
//  - V tile staged transposed Vt[64][32] so P*V B-frags are contiguous ds loads
// grid.x = BS*NHEAD, grid.y = SEQ/128; 8 waves x 16 Q rows
// ---------------------------------------------------------------------------
__global__ __launch_bounds__(256) void attn_fused(const __bf16* __restrict__ q,
                                                  const __bf16* __restrict__ k,
                                                  const __bf16* __restrict__ v,
                                                  __bf16* __restrict__ sa) {
    __shared__ __bf16 k_lds[32 * DHEAD];        // 4 KiB, row-major [n][dh]
    __shared__ __bf16 vt_lds[DHEAD * 32];       // 4 KiB, transposed [dh][m]
    __shared__ __bf16 p_lds[8][16 * 32];        // 8 KiB, per-wave P bounce

    const int bh = blockIdx.x;
    const int b_ = bh >> 3;
    const int h  = bh & 7;
    const int wave = threadIdx.x >> 5;
    const int tid  = threadIdx.x;
    const int i0 = blockIdx.y * 128 + wave * 16;

    const __bf16* Q = q + (size_t)bh * SEQ * DHEAD;
    const __bf16* K = k + (size_t)bh * SEQ * DHEAD;
    const __bf16* V = v + (size_t)bh * SEQ * DHEAD;

    // loop-invariant Q tile (16 x 64) = two A fragments
    const v16bf qa0 = load_frag_kcontig(Q + (size_t)i0 * DHEAD + 0,  DHEAD);
    const v16bf qa1 = load_frag_kcontig(Q + (size_t)i0 * DHEAD + 32, DHEAD);

    v8f o[4] = {v8f{}, v8f{}, v8f{}, v8f{}};
    float mrow[8], lrow[8];
#pragma unroll
    for (int r = 0; r < 8; ++r) { mrow[r] = -1.0e30f; lrow[r] = 0.0f; }

    const float scale = 0.125f; // 1/sqrt(64)
    const int lane = threadIdx.x & 31;
    const int ncol = lane & 15;
    const int mb   = (lane >> 4) ? 8 : 0;
    __bf16* pl = p_lds[wave];

    const unsigned klds_base = lds_offset_of(k_lds);
    const int vm = tid >> 3;          // 0..31 : V row within tile
    const int vc = (tid & 7) * 8;     // 0..56 : first of 8 dh columns

    for (int j = 0; j < SEQ; j += 32) {
        // ---- stage K tile: async global->LDS, 16B per thread (4KiB total) ----
        async_load_b128(klds_base + tid * 16, K + (size_t)j * DHEAD, tid * 16);

        // ---- stage V tile transposed: b128 global read, scattered ds stores ----
        {
            const __bf16* vsrc = V + (size_t)(j + vm) * DHEAD + vc;
#pragma unroll
            for (int i = 0; i < 8; ++i)
                vt_lds[(size_t)(vc + i) * 32 + vm] = vsrc[i];
        }

        // prefetch next tiles toward L2 while we compute this one
        if (j + 32 < SEQ) {
            __builtin_prefetch(K + (size_t)(j + 32) * DHEAD, 0, 1);
            __builtin_prefetch(V + (size_t)(j + 32) * DHEAD, 0, 1);
        }

        wait_asynccnt0();
        __syncthreads();

        // ---- scores S = Q * K^T for columns j..j+31 (B-frags from LDS) ----
        v16bf kb00 = load_frag_kcontig(k_lds + 0,                 DHEAD);
        v16bf kb01 = load_frag_kcontig(k_lds + 32,                DHEAD);
        v16bf kb10 = load_frag_kcontig(k_lds + 16 * DHEAD + 0,    DHEAD);
        v16bf kb11 = load_frag_kcontig(k_lds + 16 * DHEAD + 32,   DHEAD);

        v8f s0 = v8f{}, s1 = v8f{};
        s0 = wmma_bf16(qa0, kb00, s0);
        s0 = wmma_bf16(qa1, kb01, s0);
        s1 = wmma_bf16(qa0, kb10, s1);
        s1 = wmma_bf16(qa1, kb11, s1);

        // ---- online softmax (row stats live per 16-lane half) ----
#pragma unroll
        for (int r = 0; r < 8; ++r) {
            float x0 = s0[r] * scale;
            float x1 = s1[r] * scale;
            float tmax = fmaxf(x0, x1);
#pragma unroll
            for (int off = 1; off < 16; off <<= 1)
                tmax = fmaxf(tmax, __shfl_xor(tmax, off, 32));
            const float mnew  = fmaxf(mrow[r], tmax);
            const float alpha = __expf(mrow[r] - mnew);
            const float p0 = __expf(x0 - mnew);
            const float p1 = __expf(x1 - mnew);
            float tsum = p0 + p1;
#pragma unroll
            for (int off = 1; off < 16; off <<= 1)
                tsum += __shfl_xor(tsum, off, 32);
            lrow[r] = lrow[r] * alpha + tsum;
            mrow[r] = mnew;
            s0[r] = p0;
            s1[r] = p1;
#pragma unroll
            for (int c = 0; c < 4; ++c) o[c][r] *= alpha;
        }

        // ---- reshape P (C-layout) -> A-layout via per-wave LDS bounce ----
#pragma unroll
        for (int r = 0; r < 8; ++r) {
            pl[(size_t)(mb + r) * 32 + ncol]      = (__bf16)s0[r];
            pl[(size_t)(mb + r) * 32 + 16 + ncol] = (__bf16)s1[r];
        }
        asm volatile("s_wait_dscnt 0" ::: "memory");
        v16bf pa = load_frag_kcontig(pl, 32);

        // ---- O += P(16x32) * V(32x64); V B-frags contiguous from Vt ----
#pragma unroll
        for (int c = 0; c < 4; ++c) {
            v16bf vb = load_frag_kcontig(vt_lds + (size_t)(16 * c) * 32, 32);
            o[c] = wmma_bf16(pa, vb, o[c]);
        }

        __syncthreads();   // protect K/V tiles before next overwrite
    }

    // ---- epilogue: normalize and scatter to sa[b][i][h*64+col] ----
#pragma unroll
    for (int c = 0; c < 4; ++c) {
#pragma unroll
        for (int r = 0; r < 8; ++r) {
            const float val = o[c][r] / lrow[r];
            const int irow = i0 + mb + r;
            sa[((size_t)b_ * SEQ + irow) * DMODEL + h * DHEAD + 16 * c + ncol] = (__bf16)val;
        }
    }
}

// ---------------------------------------------------------------------------
// Output projection: out[m,n] = sum_d sa[m,d] * w_agg[n,d] + b_agg[n]
// wave tile: 32 rows x 64 cols
// ---------------------------------------------------------------------------
__global__ __launch_bounds__(256) void out_gemm(const __bf16* __restrict__ sab,
                                                const __bf16* __restrict__ wab,
                                                const float* __restrict__ bias,
                                                float* __restrict__ out) {
    const int wave = threadIdx.x >> 5;
    const int m0 = blockIdx.x * 256 + wave * 32;
    const int n0 = blockIdx.y * 64;

    v8f acc[2][4];
#pragma unroll
    for (int i = 0; i < 2; ++i)
#pragma unroll
        for (int t = 0; t < 4; ++t) acc[i][t] = v8f{};

    for (int kk = 0; kk < DMODEL; kk += 32) {
        v16bf a0 = load_frag_kcontig(sab + (size_t)m0 * DMODEL + kk, DMODEL);
        v16bf a1 = load_frag_kcontig(sab + (size_t)(m0 + 16) * DMODEL + kk, DMODEL);
#pragma unroll
        for (int t = 0; t < 4; ++t) {
            v16bf b = load_frag_kcontig(wab + (size_t)(n0 + 16 * t) * DMODEL + kk, DMODEL);
            acc[0][t] = wmma_bf16(a0, b, acc[0][t]);
            acc[1][t] = wmma_bf16(a1, b, acc[1][t]);
        }
    }

    const int lane  = threadIdx.x & 31;
    const int ncol  = lane & 15;
    const int rbase = (lane >> 4) ? 8 : 0;
#pragma unroll
    for (int i = 0; i < 2; ++i) {
#pragma unroll
        for (int t = 0; t < 4; ++t) {
#pragma unroll
            for (int r = 0; r < 8; ++r) {
                const int m = m0 + 16 * i + rbase + r;
                const int n = n0 + 16 * t + ncol;
                out[(size_t)m * DMODEL + n] = acc[i][t][r] + bias[n];
            }
        }
    }
}

// ---------------------------------------------------------------------------
// launch
// ---------------------------------------------------------------------------
extern "C" void kernel_launch(void* const* d_in, const int* in_sizes, int n_in,
                              void* d_out, int out_size, void* d_ws, size_t ws_size,
                              hipStream_t stream) {
    const float* z     = (const float*)d_in[0];   // [2,4096,512]
    const float* w_qkv = (const float*)d_in[1];   // [1536,512]
    const float* w_agg = (const float*)d_in[2];   // [512,512]
    const float* b_agg = (const float*)d_in[3];   // [512]
    float* out = (float*)d_out;                   // [2,4096,512]

    char* ws = (char*)d_ws;
    size_t off = 0;
    auto carve = [&](size_t bytes) { char* p = ws + off; off += (bytes + 255) & ~(size_t)255; return p; };

    __bf16* zb    = (__bf16*)carve((size_t)MTOT * DMODEL * 2);
    __bf16* wqkvb = (__bf16*)carve((size_t)E3 * DMODEL * 2);
    __bf16* waggb = (__bf16*)carve((size_t)DMODEL * DMODEL * 2);
    __bf16* qb    = (__bf16*)carve((size_t)BS * NHEAD * SEQ * DHEAD * 2);
    __bf16* kb    = (__bf16*)carve((size_t)BS * NHEAD * SEQ * DHEAD * 2);
    __bf16* vb    = (__bf16*)carve((size_t)BS * NHEAD * SEQ * DHEAD * 2);
    __bf16* sab   = (__bf16*)carve((size_t)MTOT * DMODEL * 2);

    cvt_f32_bf16<<<1024, 256, 0, stream>>>(z,     zb,    (long)MTOT * DMODEL);
    cvt_f32_bf16<<<512,  256, 0, stream>>>(w_qkv, wqkvb, (long)E3 * DMODEL);
    cvt_f32_bf16<<<256,  256, 0, stream>>>(w_agg, waggb, (long)DMODEL * DMODEL);

    qkv_gemm<<<dim3(MTOT / 256, E3 / 64), 256, 0, stream>>>(zb, wqkvb, qb, kb, vb);

    attn_fused<<<dim3(BS * NHEAD, SEQ / 128), 256, 0, stream>>>(qb, kb, vb, sab);

    out_gemm<<<dim3(MTOT / 256, DMODEL / 64), 256, 0, stream>>>(sab, waggb, b_agg, out);
}
